// DynamicsSolver_68504728371702
// MI455X (gfx1250) — matrix operand order repaired and checked by reading, not compiled, not run on hardware
//
#include <hip/hip_runtime.h>
#include <hip/hip_bf16.h>
#include <math.h>

typedef __attribute__((ext_vector_type(16))) _Float16 v16h;
typedef __attribute__((ext_vector_type(8)))  _Float16 v8h;
typedef __attribute__((ext_vector_type(2)))  _Float16 h2v;
typedef __attribute__((ext_vector_type(8)))  float    v8f;

#define LAT 128
#define WSTR 136   // padded LDS row stride (halves): bank-conflict-free, 16B aligned

union BU { v16h v; v8h h[2]; };

// -------- input index map (setup_inputs dict order, params leaves in insertion order)
enum {
  IN_EIDX = 0, IN_SPOS, IN_RPOS, IN_EDX, IN_EATTR,
  IN_VA, IN_VB, IN_VC, IN_SV, IN_SW, IN_RV, IN_RW,
  IN_NLAT, IN_NTYPE, IN_NW, IN_NVEL,
  P_EF_W1, P_EF_B1, P_EF_W2, P_EF_B2, P_EF_G, P_EF_BB,       // edge_feat (LN)
  P_NF_W1, P_NF_B1, P_NF_W2, P_NF_B2, P_NF_G, P_NF_BB,       // node_feat (LN)
  P_IE_W1, P_IE_B1, P_IE_W2, P_IE_B2, P_IE_G, P_IE_BB,       // inter_enc (LN)
  P_I1_W1, P_I1_B1, P_I1_W2, P_I1_B2,                        // i1
  P_I2_W1, P_I2_B1, P_I2_W2, P_I2_B2,                        // i2
  P_FS_W1, P_FS_B1, P_FS_W2, P_FS_B2,                        // f_scaler
  P_IM_W1, P_IM_B1, P_IM_W2, P_IM_B2,                        // inv_mass
  P_II_W1, P_II_B1, P_II_W2, P_II_B2,                        // inv_inertia
  P_XD_W1, P_XD_B1, P_XD_W2, P_XD_B2                         // ext_dv
};

// =====================================================================
// WMMA GEMM: h = X[M,K](f16) @ W[K,128](f32->f16) + bias
// mode 0: ReLU -> Yh/Yf            mode 1: LayerNorm(gamma,beta) -> Yh/Yf
// mode 2: ReLU then fused tiny head: Yf[M,dout] = relu(h) @ W2[128,dout] + b2
// Block: 256 thr (8 waves), 256 rows. Wave w owns row strips
// [blk*256+16w, +16) and [blk*256+128+16w, +16), all 128 cols.
// Each B fragment feeds 2 WMMAs; B loads ping-pong ahead of use.
// =====================================================================
__global__ __launch_bounds__(256) void gemm128_wmma_kernel(
    const _Float16* __restrict__ X, int ldx,
    const float* __restrict__ W,
    const float* __restrict__ bias,
    const float* __restrict__ gamma,
    const float* __restrict__ beta,
    const float* __restrict__ W2,     // mode 2 head weights [128,dout]
    const float* __restrict__ b2,     // mode 2 head bias [dout]
    int dout,
    _Float16* __restrict__ Yh, int ldyh,
    float* __restrict__ Yf, int ldyf,
    int M, int K, int mode)
{
  __shared__ __attribute__((aligned(16))) _Float16 wT[128 * WSTR]; // [n][k] padded
  __shared__ float w2s[128 * 3];
  const int tid  = threadIdx.x;
  const int wv   = tid >> 5;
  const int lane = tid & 31;
  const int half = lane >> 4;           // 0: lanes 0-15, 1: lanes 16-31
  const int ln16 = lane & 15;
  const int rowBase = blockIdx.x * 256 + wv * 16;

  if (mode == 2) {
    for (int i = tid; i < 128 * dout; i += 256) w2s[i] = W2[i];
  }

  v8f acc[2][8];
  #pragma unroll
  for (int g = 0; g < 2; ++g)
    #pragma unroll
    for (int t = 0; t < 8; ++t)
      #pragma unroll
      for (int j = 0; j < 8; ++j) acc[g][t][j] = 0.0f;

  int rA0 = rowBase + ln16;        if (rA0 > M - 1) rA0 = M - 1;
  int rA1 = rowBase + 128 + ln16;  if (rA1 > M - 1) rA1 = M - 1;

  for (int kc = 0; kc < K; kc += 128) {
    // stage weights f32->f16, transposed + padded, packed pair stores
    for (int i = tid; i < 128 * 64; i += 256) {
      int n  = i >> 6;
      int k2 = (i & 63) << 1;
      h2v p;
      p.x = (_Float16)W[(size_t)(kc + k2) * 128 + n];
      p.y = (_Float16)W[(size_t)(kc + k2 + 1) * 128 + n];
      *(h2v*)(wT + n * WSTR + k2) = p;
    }
    __syncthreads();
    const _Float16* x0 = X + (size_t)rA0 * ldx + kc + half * 8;
    const _Float16* x1 = X + (size_t)rA1 * ldx + kc + half * 8;
    #pragma unroll
    for (int s = 0; s < 4; ++s) {       // 4 x K=32 steps
      BU A0, A1;
      A0.h[0] = *(const v8h*)(x0 + s * 32);
      A0.h[1] = *(const v8h*)(x0 + s * 32 + 16);
      A1.h[0] = *(const v8h*)(x1 + s * 32);
      A1.h[1] = *(const v8h*)(x1 + s * 32 + 16);
      const int kb = s * 32 + half * 8;
      BU B[2];
      {
        const _Float16* wp = wT + ln16 * WSTR + kb;        // t = 0
        B[0].h[0] = *(const v8h*)(wp);
        B[0].h[1] = *(const v8h*)(wp + 16);
      }
      #pragma unroll
      for (int t = 0; t < 8; ++t) {
        if (t < 7) {                                        // prefetch t+1
          const _Float16* wp = wT + ((t + 1) * 16 + ln16) * WSTR + kb;
          B[(t + 1) & 1].h[0] = *(const v8h*)(wp);
          B[(t + 1) & 1].h[1] = *(const v8h*)(wp + 16);
        }
        acc[0][t] = __builtin_amdgcn_wmma_f32_16x16x32_f16(
            false, A0.v, false, B[t & 1].v, (short)0, acc[0][t], false, false);
        acc[1][t] = __builtin_amdgcn_wmma_f32_16x16x32_f16(
            false, A1.v, false, B[t & 1].v, (short)0, acc[1][t], false, false);
      }
    }
    __syncthreads();
  }

  // -------- epilogue --------
  #pragma unroll
  for (int g = 0; g < 2; ++g) {
    const int gBase = rowBase + g * 128;
    if (mode == 0) {                      // bias + ReLU
      #pragma unroll
      for (int j = 0; j < 8; ++j) {
        int row = gBase + j + half * 8;
        bool ok = row < M;
        #pragma unroll
        for (int t = 0; t < 8; ++t) {
          int n = t * 16 + ln16;
          float v = acc[g][t][j] + bias[n];
          v = v > 0.0f ? v : 0.0f;
          if (ok) {
            if (Yh) Yh[(size_t)row * ldyh + n] = (_Float16)v;
            if (Yf) Yf[(size_t)row * ldyf + n] = v;
          }
        }
      }
    } else if (mode == 1) {               // bias + LayerNorm
      #pragma unroll
      for (int j = 0; j < 8; ++j) {
        int row = gBase + j + half * 8;
        bool ok = row < M;
        float vals[8];
        float s = 0.0f, ss = 0.0f;
        #pragma unroll
        for (int t = 0; t < 8; ++t) {
          float v = acc[g][t][j] + bias[t * 16 + ln16];
          vals[t] = v; s += v; ss += v * v;
        }
        #pragma unroll
        for (int o = 1; o < 16; o <<= 1) {   // butterfly within 16-lane half
          s  += __shfl_xor(s,  o, 32);
          ss += __shfl_xor(ss, o, 32);
        }
        float mean = s * (1.0f / 128.0f);
        float var  = ss * (1.0f / 128.0f) - mean * mean;
        float inv  = rsqrtf(var + 1e-5f);
        #pragma unroll
        for (int t = 0; t < 8; ++t) {
          int n = t * 16 + ln16;
          float v = (vals[t] - mean) * inv * gamma[n] + beta[n];
          if (ok) {
            if (Yh) Yh[(size_t)row * ldyh + n] = (_Float16)v;
            if (Yf) Yf[(size_t)row * ldyf + n] = v;
          }
        }
      }
    } else {                              // mode 2: ReLU + fused tiny head
      #pragma unroll
      for (int j = 0; j < 8; ++j) {
        int row = gBase + j + half * 8;
        bool ok = row < M;
        float vals[8];
        #pragma unroll
        for (int t = 0; t < 8; ++t) {
          float v = acc[g][t][j] + bias[t * 16 + ln16];
          vals[t] = v > 0.0f ? v : 0.0f;
        }
        for (int o = 0; o < dout; ++o) {
          float p = 0.0f;
          #pragma unroll
          for (int t = 0; t < 8; ++t)
            p += vals[t] * w2s[(t * 16 + ln16) * dout + o];
          #pragma unroll
          for (int off = 1; off < 16; off <<= 1) p += __shfl_xor(p, off, 32);
          if (ok && ln16 == 0) Yf[(size_t)row * dout + o] = p + b2[o];
        }
      }
    }
  }
}

// =====================================================================
// Edge feature prep: basis projections + first (K=6 / K=4) MLP layers
// one block (128 thr) per edge
// =====================================================================
__device__ __forceinline__ float dot3(const float* a, const float* b) {
  return a[0]*b[0] + a[1]*b[1] + a[2]*b[2];
}

__global__ __launch_bounds__(128) void edge_prep_kernel(
    const float* __restrict__ va, const float* __restrict__ vb, const float* __restrict__ vc,
    const float* __restrict__ sv, const float* __restrict__ sw,
    const float* __restrict__ rv, const float* __restrict__ rw,
    const float* __restrict__ edx, const float* __restrict__ eattr,
    const float* __restrict__ w1nf, const float* __restrict__ b1nf,
    const float* __restrict__ w1ef, const float* __restrict__ b1ef,
    _Float16* __restrict__ hS, _Float16* __restrict__ hR, _Float16* __restrict__ hE,
    int E_)
{
  int e = blockIdx.x;
  if (e >= E_) return;
  __shared__ float sf[6], rf[6], ef4[4];
  if (threadIdx.x == 0) {
    float A0[3] = {va[e*3], va[e*3+1], va[e*3+2]};
    float B0[3] = {vb[e*3], vb[e*3+1], vb[e*3+2]};
    float C0[3] = {vc[e*3], vc[e*3+1], vc[e*3+2]};
    float SV[3] = {sv[e*3], sv[e*3+1], sv[e*3+2]};
    float SW[3] = {sw[e*3], sw[e*3+1], sw[e*3+2]};
    float RV[3] = {rv[e*3], rv[e*3+1], rv[e*3+2]};
    float RW[3] = {rw[e*3], rw[e*3+1], rw[e*3+2]};
    sf[0] = dot3(A0, SV); sf[1] = dot3(B0, SV); sf[2] = dot3(C0, SV);
    sf[3] = dot3(A0, SW); sf[4] = dot3(B0, SW); sf[5] = dot3(C0, SW);
    rf[0] = -dot3(A0, RV); rf[1] = -dot3(B0, RV); rf[2] = -dot3(C0, RV);
    rf[3] = -dot3(A0, RW); rf[4] = -dot3(B0, RW); rf[5] = -dot3(C0, RW);
    float d0 = edx[e*3], d1 = edx[e*3+1], d2 = edx[e*3+2];
    ef4[0] = sqrtf(d0*d0 + d1*d1 + d2*d2);
    ef4[1] = eattr[e*3]; ef4[2] = eattr[e*3+1]; ef4[3] = eattr[e*3+2];
  }
  __syncthreads();
  int j = threadIdx.x;
  float as = b1nf[j], ar = b1nf[j], ae = b1ef[j];
  #pragma unroll
  for (int k = 0; k < 6; ++k) {
    float wkj = w1nf[k * LAT + j];
    as += sf[k] * wkj;
    ar += rf[k] * wkj;
  }
  #pragma unroll
  for (int k = 0; k < 4; ++k) ae += ef4[k] * w1ef[k * LAT + j];
  size_t o = (size_t)e * LAT + j;
  hS[o] = (_Float16)(as > 0.0f ? as : 0.0f);
  hR[o] = (_Float16)(ar > 0.0f ? ar : 0.0f);
  hE[o] = (_Float16)(ae > 0.0f ? ae : 0.0f);
}

// Build inter_in cols 0..255 (cols 256..383 written by e_lat GEMM directly)
__global__ __launch_bounds__(256) void inter_build_kernel(
    const _Float16* __restrict__ sLat, const _Float16* __restrict__ rLat,
    const float* __restrict__ nlat, const long long* __restrict__ eidx,
    _Float16* __restrict__ H, int E_)
{
  int e = blockIdx.x, j = threadIdx.x;          // j in [0,256)
  if (e >= E_) return;
  _Float16 v;
  if (j < 128) {
    v = (_Float16)((float)sLat[(size_t)e * LAT + j] + (float)rLat[(size_t)e * LAT + j]);
  } else {
    int jj = j - 128;
    long long s = eidx[e], r = eidx[E_ + e];
    v = (_Float16)(nlat[(size_t)s * LAT + jj] + nlat[(size_t)r * LAT + jj]);
  }
  H[(size_t)e * 384 + j] = v;
}

// f32 -> f16 copy (node_latent)
__global__ void cvt_f16_kernel(const float* __restrict__ x, _Float16* __restrict__ y, size_t n) {
  size_t i = (size_t)blockIdx.x * blockDim.x + threadIdx.x;
  if (i < n) y[i] = (_Float16)x[i];
}

__global__ void zero_kernel(float* __restrict__ p, size_t n) {
  size_t i = (size_t)blockIdx.x * blockDim.x + threadIdx.x;
  if (i < n) p[i] = 0.0f;
}

// pass 1: mask, masked COM mass sums, fij/aij and their masked group sums
__global__ void edge_pass1_kernel(
    const long long* __restrict__ eidx, const float* __restrict__ eattr,
    const int* __restrict__ ntype, const float* __restrict__ nwts,
    const float* __restrict__ dvRaw, const float* __restrict__ cf, const float* __restrict__ ca,
    const float* __restrict__ va, const float* __restrict__ vb, const float* __restrict__ vc,
    float* __restrict__ wmask, float* __restrict__ fij, float* __restrict__ aij,
    float* __restrict__ mSum, float* __restrict__ cnt,
    float* __restrict__ mdv, float* __restrict__ fSum, float* __restrict__ aSum, int E_)
{
  int e = blockIdx.x * blockDim.x + threadIdx.x;
  if (e >= E_) return;
  long long s = eidx[e], r = eidx[E_ + e];
  bool isv = (eattr[e*3] == -1.0f);
  bool gr  = (ntype[r*2 + 1] == -1);
  bool gs  = (ntype[s*2 + 1] == -1);
  float w = (isv && gr && !gs) ? 1.0f : 0.0f;
  wmask[e] = w;
  float c0 = cf[e*3], c1 = cf[e*3+1], c2 = cf[e*3+2];
  float d0 = ca[e*3], d1 = ca[e*3+1], d2 = ca[e*3+2];
  float f[3], a[3];
  #pragma unroll
  for (int d = 0; d < 3; ++d) {
    float A0 = va[e*3+d], B0 = vb[e*3+d], C0 = vc[e*3+d];
    f[d] = c0*A0 + c1*B0 + c2*C0;
    a[d] = d0*A0 + d1*B0 + d2*C0;
    fij[e*3+d] = f[d];
    aij[e*3+d] = a[d];
  }
  if (w != 0.0f) {
    float m = fmaxf(nwts[s], 1e-12f);
    atomicAdd(&mSum[r], m);
    atomicAdd(&cnt[r], 1.0f);
    #pragma unroll
    for (int d = 0; d < 3; ++d) {
      atomicAdd(&mdv[r*3+d], dvRaw[s*3+d] * m);
      atomicAdd(&fSum[r*3+d], f[d]);
      atomicAdd(&aSum[r*3+d], a[d]);
    }
  }
}

__global__ void node_mid_kernel(
    const float* __restrict__ mSum, const float* __restrict__ cnt,
    const float* __restrict__ mdv, const float* __restrict__ fSum, const float* __restrict__ aSum,
    float* __restrict__ dvCom, float* __restrict__ fMean, float* __restrict__ aMean, int N_)
{
  int n = blockIdx.x * blockDim.x + threadIdx.x;
  if (n >= N_) return;
  float Mg = fmaxf(mSum[n], 1e-12f);
  float c  = fmaxf(cnt[n], 1.0f);
  #pragma unroll
  for (int d = 0; d < 3; ++d) {
    dvCom[n*3+d] = mdv[n*3+d] / Mg;
    fMean[n*3+d] = fSum[n*3+d] / c;
    aMean[n*3+d] = aSum[n*3+d] / c;
  }
}

// pass 2: group-mean removal, moment, scatter delta / net force / net torque
__global__ void edge_pass2_kernel(
    const long long* __restrict__ eidx, const float* __restrict__ wmask,
    const float* __restrict__ dvRaw, const float* __restrict__ dvCom,
    const float* __restrict__ fij, const float* __restrict__ aij,
    const float* __restrict__ fMean, const float* __restrict__ aMean,
    const float* __restrict__ lam, const float* __restrict__ spos, const float* __restrict__ rpos,
    float* __restrict__ dAcc, float* __restrict__ nF, float* __restrict__ nT, int E_)
{
  int e = blockIdx.x * blockDim.x + threadIdx.x;
  if (e >= E_) return;
  long long s = eidx[e], r = eidx[E_ + e];
  float w = wmask[e];
  float f[3], a[3], lev[3];
  #pragma unroll
  for (int d = 0; d < 3; ++d) {
    f[d]  = fij[e*3+d] - fMean[r*3+d] * w;
    a[d]  = aij[e*3+d] - aMean[r*3+d] * w;
    lev[d] = spos[e*3+d] - rpos[e*3+d];
  }
  if (w != 0.0f) {
    #pragma unroll
    for (int d = 0; d < 3; ++d)
      atomicAdd(&dAcc[s*3+d], dvRaw[r*3+d] - dvCom[r*3+d]);
  }
  float l = lam[e];
  float fl[3] = {f[0]*l, f[1]*l, f[2]*l};
  float mom[3] = {lev[1]*fl[2] - lev[2]*fl[1],
                  lev[2]*fl[0] - lev[0]*fl[2],
                  lev[0]*fl[1] - lev[1]*fl[0]};
  #pragma unroll
  for (int d = 0; d < 3; ++d) {
    atomicAdd(&nF[r*3+d], f[d]);
    atomicAdd(&nT[r*3+d], a[d] - mom[d]);
  }
}

__global__ void node_final_kernel(
    const float* __restrict__ dvRaw, const float* __restrict__ dAcc,
    const float* __restrict__ invM, const float* __restrict__ invI,
    const float* __restrict__ nF, const float* __restrict__ nT,
    float* __restrict__ outDv, float* __restrict__ outDw, int N_)
{
  int n = blockIdx.x * blockDim.x + threadIdx.x;
  if (n >= N_) return;
  #pragma unroll
  for (int d = 0; d < 3; ++d) {
    outDv[n*3+d] = invM[n] * nF[n*3+d] + dvRaw[n*3+d] + dAcc[n*3+d];
    outDw[n*3+d] = invI[n] * nT[n*3+d];
  }
}

// =====================================================================
extern "C" void kernel_launch(void* const* d_in, const int* in_sizes, int n_in,
                              void* d_out, int out_size, void* d_ws, size_t ws_size,
                              hipStream_t stream) {
  const int E_ = in_sizes[IN_EATTR] / 3;
  const int N_ = in_sizes[IN_NLAT] / LAT;

  const long long* eidx = (const long long*)d_in[IN_EIDX];
  const float* spos  = (const float*)d_in[IN_SPOS];
  const float* rpos  = (const float*)d_in[IN_RPOS];
  const float* edx   = (const float*)d_in[IN_EDX];
  const float* eattr = (const float*)d_in[IN_EATTR];
  const float* va = (const float*)d_in[IN_VA];
  const float* vb = (const float*)d_in[IN_VB];
  const float* vc = (const float*)d_in[IN_VC];
  const float* sv = (const float*)d_in[IN_SV];
  const float* sw = (const float*)d_in[IN_SW];
  const float* rv = (const float*)d_in[IN_RV];
  const float* rw = (const float*)d_in[IN_RW];
  const float* nlat  = (const float*)d_in[IN_NLAT];
  const int*   ntype = (const int*)d_in[IN_NTYPE];
  const float* nwts  = (const float*)d_in[IN_NW];
  #define P(i) ((const float*)d_in[i])

  float* out    = (float*)d_out;
  float* outDv  = out;
  float* outDw  = out + (size_t)N_ * 3;
  float* outIL  = out + (size_t)N_ * 6;

  // ---- workspace carve ----
  size_t off = 0;
  auto alloc = [&](size_t bytes) -> void* {
    void* p = (char*)d_ws + off;
    off += (bytes + 255) & ~(size_t)255;
    return p;
  };
  const size_t EL = (size_t)E_ * LAT;
  // bufA/bufB contiguous (combined s/r GEMM input), bufD/bufF contiguous (its output)
  _Float16* bufA = (_Float16*)alloc(EL * 2);       // h_s   (EL*2 is 256-aligned)
  _Float16* bufB = (_Float16*)alloc(EL * 2);       // h_r == bufA + EL
  _Float16* bufC = (_Float16*)alloc(EL * 2);       // h_e
  _Float16* bufD = (_Float16*)alloc(EL * 2);       // s_lat -> h_ie
  _Float16* bufF = (_Float16*)alloc(EL * 2);       // r_lat -> inter_lat f16 == bufD + EL
  _Float16* bufH = (_Float16*)alloc((size_t)E_ * 384 * 2);  // inter_in (e_lat in cols 256+)
  _Float16* nl16 = (_Float16*)alloc((size_t)N_ * LAT * 2);
  float* coefF = (float*)alloc((size_t)E_ * 3 * 4);
  float* coefA = (float*)alloc((size_t)E_ * 3 * 4);
  float* lam   = (float*)alloc((size_t)E_ * 4);
  float* wmask = (float*)alloc((size_t)E_ * 4);
  float* fij   = (float*)alloc((size_t)E_ * 3 * 4);
  float* aij   = (float*)alloc((size_t)E_ * 3 * 4);
  float* dvRaw = (float*)alloc((size_t)N_ * 3 * 4);
  float* invM  = (float*)alloc((size_t)N_ * 4);
  float* invI  = (float*)alloc((size_t)N_ * 4);
  float* accum = (float*)alloc((size_t)N_ * 20 * 4);  // zeroed block
  float* mSum = accum;              float* cnt  = accum + N_;
  float* mdv  = accum + 2*(size_t)N_;  float* fSum = accum + 5*(size_t)N_;
  float* aSum = accum + 8*(size_t)N_;  float* dAcc = accum + 11*(size_t)N_;
  float* nF   = accum + 14*(size_t)N_; float* nT   = accum + 17*(size_t)N_;
  float* dvCom = (float*)alloc((size_t)N_ * 3 * 4);
  float* fMean = (float*)alloc((size_t)N_ * 3 * 4);
  float* aMean = (float*)alloc((size_t)N_ * 3 * 4);
  (void)ws_size; (void)n_in; (void)out_size;

  const int gE  = (E_ + 255) / 256;        // gemm blocks over edges
  const int gE2 = (2 * E_ + 255) / 256;    // combined s/r gemm
  const int gN  = (N_ + 255) / 256;        // gemm blocks over nodes

  // 1) first layers of the three encoders
  edge_prep_kernel<<<E_, 128, 0, stream>>>(va, vb, vc, sv, sw, rv, rw, edx, eattr,
      P(P_NF_W1), P(P_NF_B1), P(P_EF_W1), P(P_EF_B1), bufA, bufB, bufC, E_);

  // 2) second layers + LN:  s_lat & r_lat fused (same weights, contiguous buffers),
  //    e_lat straight into inter_in cols 256+
  gemm128_wmma_kernel<<<gE2, 256, 0, stream>>>(bufA, LAT, P(P_NF_W2), P(P_NF_B2),
      P(P_NF_G), P(P_NF_BB), nullptr, nullptr, 0, bufD, LAT, nullptr, 0, 2 * E_, LAT, 1);
  gemm128_wmma_kernel<<<gE, 256, 0, stream>>>(bufC, LAT, P(P_EF_W2), P(P_EF_B2),
      P(P_EF_G), P(P_EF_BB), nullptr, nullptr, 0, bufH + 256, 384, nullptr, 0, E_, LAT, 1);

  // 3) inter_in cols 0..255
  inter_build_kernel<<<E_, 256, 0, stream>>>(bufD, bufF, nlat, eidx, bufH, E_);

  // 4) inter_enc layer1 (K=384) -> h_ie ; layer2 + LN -> interaction_latent (f16 + f32 out)
  gemm128_wmma_kernel<<<gE, 256, 0, stream>>>(bufH, 384, P(P_IE_W1), P(P_IE_B1),
      nullptr, nullptr, nullptr, nullptr, 0, bufD, LAT, nullptr, 0, E_, 384, 0);
  gemm128_wmma_kernel<<<gE, 256, 0, stream>>>(bufD, LAT, P(P_IE_W2), P(P_IE_B2),
      P(P_IE_G), P(P_IE_BB), nullptr, nullptr, 0, bufF, LAT, outIL, LAT, E_, LAT, 1);

  // 5) decoder heads fused into the GEMM epilogue (mode 2)
  gemm128_wmma_kernel<<<gE, 256, 0, stream>>>(bufF, LAT, P(P_I1_W1), P(P_I1_B1),
      nullptr, nullptr, P(P_I1_W2), P(P_I1_B2), 3, nullptr, 0, coefF, 0, E_, LAT, 2);
  gemm128_wmma_kernel<<<gE, 256, 0, stream>>>(bufF, LAT, P(P_I2_W1), P(P_I2_B1),
      nullptr, nullptr, P(P_I2_W2), P(P_I2_B2), 3, nullptr, 0, coefA, 0, E_, LAT, 2);
  gemm128_wmma_kernel<<<gE, 256, 0, stream>>>(bufF, LAT, P(P_FS_W1), P(P_FS_B1),
      nullptr, nullptr, P(P_FS_W2), P(P_FS_B2), 1, nullptr, 0, lam, 0, E_, LAT, 2);

  // 6) node MLPs, heads fused: ext_dv, inv_mass, inv_inertia
  cvt_f16_kernel<<<((size_t)N_*LAT + 255)/256, 256, 0, stream>>>(nlat, nl16, (size_t)N_ * LAT);
  gemm128_wmma_kernel<<<gN, 256, 0, stream>>>(nl16, LAT, P(P_XD_W1), P(P_XD_B1),
      nullptr, nullptr, P(P_XD_W2), P(P_XD_B2), 3, nullptr, 0, dvRaw, 0, N_, LAT, 2);
  gemm128_wmma_kernel<<<gN, 256, 0, stream>>>(nl16, LAT, P(P_IM_W1), P(P_IM_B1),
      nullptr, nullptr, P(P_IM_W2), P(P_IM_B2), 1, nullptr, 0, invM, 0, N_, LAT, 2);
  gemm128_wmma_kernel<<<gN, 256, 0, stream>>>(nl16, LAT, P(P_II_W1), P(P_II_B1),
      nullptr, nullptr, P(P_II_W2), P(P_II_B2), 1, nullptr, 0, invI, 0, N_, LAT, 2);

  // 7) graph part
  zero_kernel<<<((size_t)N_*20 + 255)/256, 256, 0, stream>>>(accum, (size_t)N_ * 20);
  edge_pass1_kernel<<<(E_ + 255)/256, 256, 0, stream>>>(eidx, eattr, ntype, nwts, dvRaw,
      coefF, coefA, va, vb, vc, wmask, fij, aij, mSum, cnt, mdv, fSum, aSum, E_);
  node_mid_kernel<<<(N_ + 255)/256, 256, 0, stream>>>(mSum, cnt, mdv, fSum, aSum,
      dvCom, fMean, aMean, N_);
  edge_pass2_kernel<<<(E_ + 255)/256, 256, 0, stream>>>(eidx, wmask, dvRaw, dvCom,
      fij, aij, fMean, aMean, lam, spos, rpos, dAcc, nF, nT, E_);
  node_final_kernel<<<(N_ + 255)/256, 256, 0, stream>>>(dvRaw, dAcc, invM, invI,
      nF, nT, outDv, outDw, N_);
  #undef P
}
